// SemiConv_41351945126305
// MI455X (gfx1250) — matrix operand order repaired and checked by reading, not compiled, not run on hardware
//
#include <hip/hip_runtime.h>
#include <hip/hip_bf16.h>

typedef __attribute__((ext_vector_type(16))) __bf16 v16bf;
typedef __attribute__((ext_vector_type(8)))  float  v8f;

#define HID 64
#define MPD 192
#define WPB 4            // waves per block
#define TPB (WPB * 32)
#define PAD 68           // LDS row stride (floats): 16B-aligned rows, 4-bank row shift

// ---- ISA A-fragment K index for 16-bit A 16x32 (CDNA5 ISA 7.12.2) ----
// lanes 0-15 : elems 0..7 -> K 0..7,  elems 8..15 -> K 16..23
// lanes 16-31: elems 0..7 -> K 8..15, elems 8..15 -> K 24..31
__device__ __forceinline__ int kk_of(int lane, int j) {
  return (j & 7) + ((j >> 3) << 4) + ((lane & 16) >> 1);
}

// Fragment-ready weights: Wf[((kc*4 + nc)*32 + lane)] is the lane's v16bf B fragment.
__device__ __forceinline__ v16bf load_bfrag(const v16bf* __restrict__ Wf,
                                            int kc, int nc, int lane) {
  return Wf[((kc * 4 + nc) << 5) + lane];
}

// sign-aware float atomic max (int/uint ordering trick; buffer initialized to -inf)
__device__ __forceinline__ void atomicMaxF(float* addr, float v) {
  if (v >= 0.0f) atomicMax((int*)addr, __float_as_int(v));
  else           atomicMin((unsigned int*)addr, __float_as_uint(v));
}

// Linear(192->64) + ReLU + Linear(64->64) + bias; pre-LayerNorm result left in hbuf.
// A: 16x192 bf16 in 6 fragments. Per-wave; hbuf is this wave's [16][PAD] LDS tile.
__device__ __forceinline__ void mlp_core(const v16bf A[6],
    const v16bf* __restrict__ W1f, const float* __restrict__ b1,
    const v16bf* __restrict__ W2f, const float* __restrict__ b2,
    float (*hbuf)[PAD], int lane)
{
  const int mofs = (lane & 16) ? 8 : 0;   // C/D layout: lanes>=16 hold M=8..15
  const int nlo  = lane & 15;
  // ---- GEMM1: [16,192] @ [192,64] + b1, ReLU ----
#pragma unroll
  for (int nc = 0; nc < 4; ++nc) {
    const float bias = b1[nc * 16 + nlo];
    v8f acc = {bias, bias, bias, bias, bias, bias, bias, bias};
#pragma unroll
    for (int kc = 0; kc < 6; ++kc) {
      v16bf B = load_bfrag(W1f, kc, nc, lane);
      acc = __builtin_amdgcn_wmma_f32_16x16x32_bf16(
          false, A[kc], false, B, (short)0, acc, false, false);
    }
#pragma unroll
    for (int i = 0; i < 8; ++i)
      hbuf[i + mofs][nc * 16 + nlo] = fmaxf(acc[i], 0.0f);
  }
  // ---- re-layout hbuf rows into A fragments for GEMM2 (K = 64) ----
  v16bf A2[2];
#pragma unroll
  for (int c = 0; c < 2; ++c)
#pragma unroll
    for (int j = 0; j < 16; ++j)
      A2[c][j] = (__bf16)hbuf[nlo][c * 32 + kk_of(lane, j)];
  // ---- GEMM2: [16,64] @ [64,64] + b2 (pre-LN) ----
#pragma unroll
  for (int nc = 0; nc < 4; ++nc) {
    const float bias = b2[nc * 16 + nlo];
    v8f acc = {bias, bias, bias, bias, bias, bias, bias, bias};
#pragma unroll
    for (int kc = 0; kc < 2; ++kc) {
      v16bf B = load_bfrag(W2f, kc, nc, lane);
      acc = __builtin_amdgcn_wmma_f32_16x16x32_bf16(
          false, A2[kc], false, B, (short)0, acc, false, false);
    }
#pragma unroll
    for (int i = 0; i < 8; ++i)
      hbuf[i + mofs][nc * 16 + nlo] = acc[i];
  }
}

__device__ __forceinline__ void ln_stats(float (*hbuf)[PAD], float* stat, int lane) {
  if (lane < 16) {
    float s = 0.0f, s2 = 0.0f;
#pragma unroll
    for (int f = 0; f < HID; ++f) { float v = hbuf[lane][f]; s += v; s2 += v * v; }
    const float mu  = s * (1.0f / HID);
    const float var = fmaxf(s2 * (1.0f / HID) - mu * mu, 0.0f);
    stat[lane]      = mu;
    stat[16 + lane] = rsqrtf(var + 1e-5f);
  }
}

// ---------------- edge update: xe += MLP([mean,max,sqmean](xv[src],xv[dst])) ----------------
// Also feeds the e2v segment reductions (sum / sumsq / max) with the NEW xe via atomics.
__global__ __launch_bounds__(TPB) void edge_update_kernel(
    const float* __restrict__ xv, const float* __restrict__ xe_in,
    float* __restrict__ xe_out,
    const int* __restrict__ src, const int* __restrict__ dst,
    const v16bf* __restrict__ W1f, const float* __restrict__ b1,
    const v16bf* __restrict__ W2f, const float* __restrict__ b2,
    const float* __restrict__ g, const float* __restrict__ beta,
    float* __restrict__ nsum, float* __restrict__ nsq, float* __restrict__ nmax,
    long nTiles, long E)
{
  __shared__ float aBuf[WPB][16][PAD];
  __shared__ float bBuf[WPB][16][PAD];
  __shared__ float hBuf[WPB][16][PAD];
  __shared__ float stat[WPB][32];

  const int  lane = threadIdx.x & 31;
  const int  wv   = threadIdx.x >> 5;
  const long tile = (long)blockIdx.x * WPB + wv;
  if (tile >= nTiles) return;
  const long e0 = tile * 16;

  // stage endpoint feature rows
  for (int idx = lane; idx < 16 * HID; idx += 32) {
    const int r = idx >> 6, f = idx & 63;
    const long e = e0 + r;
    float av = 0.0f, bv = 0.0f;
    if (e < E) {
      av = xv[(long)src[e] * HID + f];
      bv = xv[(long)dst[e] * HID + f];
    }
    aBuf[wv][r][f] = av;
    bBuf[wv][r][f] = bv;
  }

  // build A = [mean | max | sqmean] as bf16 fragments
  const int row = lane & 15;
  v16bf A[6];
#pragma unroll
  for (int c = 0; c < 6; ++c) {
#pragma unroll
    for (int j = 0; j < 16; ++j) {
      const int K = c * 32 + kk_of(lane, j);
      const int sel = K >> 6, f = K & 63;
      const float a = aBuf[wv][row][f], b = bBuf[wv][row][f];
      const float v = (sel == 0) ? 0.5f * (a + b)
                    : (sel == 1) ? fmaxf(a, b)
                                 : 0.5f * (a * a + b * b);
      A[c][j] = (__bf16)v;
    }
  }

  mlp_core(A, W1f, b1, W2f, b2, hBuf[wv], lane);
  ln_stats(hBuf[wv], stat[wv], lane);

  for (int idx = lane; idx < 16 * HID; idx += 32) {
    const int r = idx >> 6, f = idx & 63;
    const long e = e0 + r;
    if (e >= E) continue;
    const float h = hBuf[wv][r][f];
    float v = (h - stat[wv][r]) * stat[wv][16 + r] * g[f] + beta[f];
    v = fmaxf(v, 0.0f);
    const float nx = xe_in[e * HID + f] + v;
    xe_out[e * HID + f] = nx;
    const long so   = (long)src[e] * HID + f;
    const long dofs = (long)dst[e] * HID + f;
    atomicAdd(&nsum[so], nx);
    atomicAdd(&nsum[dofs], nx);
    const float q = nx * nx;
    atomicAdd(&nsq[so], q);
    atomicAdd(&nsq[dofs], q);
    atomicMaxF(&nmax[so], nx);
    atomicMaxF(&nmax[dofs], nx);
  }
}

// ---------------- node update: xv += MLP([sum/cnt | max | sumsq/cnt]) ----------------
__global__ __launch_bounds__(TPB) void node_update_kernel(
    const float* __restrict__ xv_in, float* __restrict__ xv_out,
    const float* __restrict__ nsum, const float* __restrict__ nsq,
    const float* __restrict__ nmax, const float* __restrict__ ncnt,
    const v16bf* __restrict__ W1f, const float* __restrict__ b1,
    const v16bf* __restrict__ W2f, const float* __restrict__ b2,
    const float* __restrict__ g, const float* __restrict__ beta,
    long nTiles, long V)
{
  __shared__ float hBuf[WPB][16][PAD];
  __shared__ float stat[WPB][32];

  const int  lane = threadIdx.x & 31;
  const int  wv   = threadIdx.x >> 5;
  const long tile = (long)blockIdx.x * WPB + wv;
  if (tile >= nTiles) return;
  const long n0 = tile * 16;

  const int row = lane & 15;
  long node = n0 + row; if (node >= V) node = V - 1;
  const float cnt = ncnt[node];
  const float inv = 1.0f / fmaxf(cnt, 1.0f);

  v16bf A[6];
#pragma unroll
  for (int c = 0; c < 6; ++c) {
#pragma unroll
    for (int j = 0; j < 16; ++j) {
      const int K = c * 32 + kk_of(lane, j);
      const int sel = K >> 6, f = K & 63;
      float v;
      if (sel == 0)      v = nsum[node * HID + f] * inv;
      else if (sel == 1) v = (cnt > 0.0f) ? nmax[node * HID + f] : 0.0f;
      else               v = nsq[node * HID + f] * inv;
      A[c][j] = (__bf16)v;
    }
  }

  mlp_core(A, W1f, b1, W2f, b2, hBuf[wv], lane);
  ln_stats(hBuf[wv], stat[wv], lane);

  for (int idx = lane; idx < 16 * HID; idx += 32) {
    const int r = idx >> 6, f = idx & 63;
    const long n = n0 + r;
    if (n >= V) continue;
    const float h = hBuf[wv][r][f];
    float v = (h - stat[wv][r]) * stat[wv][16 + r] * g[f] + beta[f];
    v = fmaxf(v, 0.0f);
    xv_out[n * HID + f] = xv_in[n * HID + f] + v;
  }
}

// ---------------- utility kernels ----------------
// Shuffle row-major f32 W[K][64] into fragment-ready bf16:
// out[((kc*4 + nc)*32 + lane)*16 + j] = W[(kc*32 + ((lane&16)?8:0) + (j&7) + ((j>>3)<<4))*64 + nc*16 + (lane&15)]
__global__ void pack_bfrag_kernel(const float* __restrict__ W, __bf16* __restrict__ out,
                                  int KC) {
  const int i = blockIdx.x * blockDim.x + threadIdx.x;
  const int total = KC << 11;          // KC * 4 * 32 * 16
  if (i >= total) return;
  const int j    = i & 15;
  const int lane = (i >> 4) & 31;
  const int nc   = (i >> 9) & 3;
  const int kc   = i >> 11;
  const int col  = nc * 16 + (lane & 15);
  const int k    = kc * 32 + ((lane & 16) ? 8 : 0) + (j & 7) + ((j >> 3) << 4);
  out[i] = (__bf16)W[k * HID + col];
}
__global__ void fill_f32_kernel(float* __restrict__ p, float v, long n) {
  const long i = (long)blockIdx.x * blockDim.x + threadIdx.x;
  if (i < n) p[i] = v;
}
__global__ void count_kernel(const int* __restrict__ src, const int* __restrict__ dst,
                             float* __restrict__ cnt, long E) {
  const long i = (long)blockIdx.x * blockDim.x + threadIdx.x;
  if (i < E) {
    atomicAdd(&cnt[src[i]], 1.0f);
    atomicAdd(&cnt[dst[i]], 1.0f);
  }
}

extern "C" void kernel_launch(void* const* d_in, const int* in_sizes, int n_in,
                              void* d_out, int out_size, void* d_ws, size_t ws_size,
                              hipStream_t stream) {
  const long E = (long)in_sizes[14];            // edge_src
  const long V = (long)in_sizes[1] / HID;       // xv rows

  const float* xe      = (const float*)d_in[0];
  const float* xv      = (const float*)d_in[1];
  const float* line_W1 = (const float*)d_in[2];
  const float* line_b1 = (const float*)d_in[3];
  const float* line_W2 = (const float*)d_in[4];
  const float* line_b2 = (const float*)d_in[5];
  const float* line_g  = (const float*)d_in[6];
  const float* line_be = (const float*)d_in[7];
  const float* linv_W1 = (const float*)d_in[8];
  const float* linv_b1 = (const float*)d_in[9];
  const float* linv_W2 = (const float*)d_in[10];
  const float* linv_b2 = (const float*)d_in[11];
  const float* linv_g  = (const float*)d_in[12];
  const float* linv_be = (const float*)d_in[13];
  const int*   src     = (const int*)d_in[14];
  const int*   dst     = (const int*)d_in[15];

  float* out_xe = (float*)d_out;
  float* out_xv = out_xe + E * HID;

  // carve workspace (256B aligned chunks)
  char*  ws   = (char*)d_ws;
  size_t used = 0;
  auto alloc = [&](size_t bytes) -> char* {
    char* p = ws + used;
    used += (bytes + 255) & ~(size_t)255;
    return p;
  };
  __bf16* W1e  = (__bf16*)alloc((size_t)6 * 2048 * sizeof(__bf16));  // 6 k-chunks
  __bf16* W2e  = (__bf16*)alloc((size_t)2 * 2048 * sizeof(__bf16));  // 2 k-chunks
  __bf16* W1v  = (__bf16*)alloc((size_t)6 * 2048 * sizeof(__bf16));
  __bf16* W2v  = (__bf16*)alloc((size_t)2 * 2048 * sizeof(__bf16));
  float*  nsum = (float*)alloc((size_t)V * HID * sizeof(float));
  float*  nsq  = (float*)alloc((size_t)V * HID * sizeof(float));
  float*  nmax = (float*)alloc((size_t)V * HID * sizeof(float));
  float*  ncnt = (float*)alloc((size_t)V * sizeof(float));
  if (used > ws_size) return;   // workspace too small: bail deterministically

  const int  T = 256;
  auto blks = [&](long n) { return (unsigned)((n + T - 1) / T); };

  // weights -> fragment-ready bf16 (once per launch; cheap)
  pack_bfrag_kernel<<<blks(6 * 2048), T, 0, stream>>>(line_W1, W1e, 6);
  pack_bfrag_kernel<<<blks(2 * 2048), T, 0, stream>>>(line_W2, W2e, 2);
  pack_bfrag_kernel<<<blks(6 * 2048), T, 0, stream>>>(linv_W1, W1v, 6);
  pack_bfrag_kernel<<<blks(2 * 2048), T, 0, stream>>>(linv_W2, W2v, 2);

  // per-node incidence counts (constant across both rounds)
  fill_f32_kernel<<<blks(V), T, 0, stream>>>(ncnt, 0.0f, V);
  count_kernel<<<blks(E), T, 0, stream>>>(src, dst, ncnt, E);

  const long eTiles  = (E + 15) / 16;
  const long vTiles  = (V + 15) / 16;
  const unsigned eBlocks = (unsigned)((eTiles + WPB - 1) / WPB);
  const unsigned vBlocks = (unsigned)((vTiles + WPB - 1) / WPB);
  const float NEG_INF = -__builtin_inff();

  // -------- round 1 --------
  fill_f32_kernel<<<blks(V * HID), T, 0, stream>>>(nsum, 0.0f, V * HID);
  fill_f32_kernel<<<blks(V * HID), T, 0, stream>>>(nsq, 0.0f, V * HID);
  fill_f32_kernel<<<blks(V * HID), T, 0, stream>>>(nmax, NEG_INF, V * HID);
  edge_update_kernel<<<eBlocks, TPB, 0, stream>>>(
      xv, xe, out_xe, src, dst,
      (const v16bf*)W1e, line_b1, (const v16bf*)W2e, line_b2, line_g, line_be,
      nsum, nsq, nmax, eTiles, E);
  node_update_kernel<<<vBlocks, TPB, 0, stream>>>(
      xv, out_xv, nsum, nsq, nmax, ncnt,
      (const v16bf*)W1v, linv_b1, (const v16bf*)W2v, linv_b2,
      linv_g, linv_be, vTiles, V);

  // -------- round 2 --------
  fill_f32_kernel<<<blks(V * HID), T, 0, stream>>>(nsum, 0.0f, V * HID);
  fill_f32_kernel<<<blks(V * HID), T, 0, stream>>>(nsq, 0.0f, V * HID);
  fill_f32_kernel<<<blks(V * HID), T, 0, stream>>>(nmax, NEG_INF, V * HID);
  edge_update_kernel<<<eBlocks, TPB, 0, stream>>>(
      out_xv, out_xe, out_xe, src, dst,
      (const v16bf*)W1e, line_b1, (const v16bf*)W2e, line_b2, line_g, line_be,
      nsum, nsq, nmax, eTiles, E);
  node_update_kernel<<<vBlocks, TPB, 0, stream>>>(
      out_xv, out_xv, nsum, nsq, nmax, ncnt,
      (const v16bf*)W1v, linv_b1, (const v16bf*)W2v, linv_b2,
      linv_g, linv_be, vTiles, V);
}
